// SelectiveSSM_1382979469550
// MI455X (gfx1250) — matrix operand order repaired and checked by reading, not compiled, not run on hardware
//
#include <hip/hip_runtime.h>
#include <hip/hip_bf16.h>

// ---------------------------------------------------------------------------
// Problem constants (from reference setup_inputs)
// ---------------------------------------------------------------------------
#define BSZ     4
#define LSEQ    4096
#define DMODEL  1024
#define DINNER  2048
#define DSTATE  16
#define DCONV   4
#define NROWS   (BSZ * LSEQ)        // 16384
#define N_XZ    (2 * DINNER)        // 4096
#define N_SSM   (2 * DSTATE + 1)    // 33

typedef __bf16 bf16_t;
typedef __attribute__((ext_vector_type(16))) __bf16 v16bf;
typedef __attribute__((ext_vector_type(8)))  float  v8f;

union FragU { v16bf v; uint4 q[2]; };

// ---------------------------------------------------------------------------
// fp32 -> bf16 conversion (memory-bound prolog)
// ---------------------------------------------------------------------------
__global__ void cvt_f32_bf16(const float* __restrict__ src,
                             bf16_t* __restrict__ dst, int n) {
    int i = blockIdx.x * blockDim.x + threadIdx.x;
    if (i < n) dst[i] = (bf16_t)src[i];
}

// ---------------------------------------------------------------------------
// WMMA bf16 GEMM:  C[M,N] = A[M,K] * B[N,K]^T   (A,B row-major bf16, C f32)
// Block = 256 threads = 8 waves arranged 4(M) x 2(N); block tile 128 x 128.
// Each wave computes a 32(M) x 64(N) tile: 2 A-frags x 4 B-frags
// -> 8 v_wmma_f32_16x16x32_bf16 per K-step (~21.3 flop/byte of L2 traffic).
// Explicit register double-buffering: k+32 fragments are loaded BEFORE the
// WMMAs on k's fragments execute, so the L2 latency of all 6 loads is hidden
// behind 8 WMMAs and s_wait_loadcnt decouples from the WMMA burst.
// Fragment layouts follow CDNA5 ISA 7.12.2 exactly (16-bit A 16x32, C 16x16 f32).
// ---------------------------------------------------------------------------
__device__ __forceinline__ v16bf load_a_frag(const bf16_t* A, int lda,
                                             int m0, int k0, int lane) {
    // lanes 0-15: row = m0+lane, K = k0+[0..7] and k0+[16..23]
    // lanes 16-31: row = m0+lane-16, K = k0+[8..15] and k0+[24..31]
    int row = m0 + (lane & 15);
    int kk  = k0 + ((lane >> 4) << 3);
    const bf16_t* p = A + (size_t)row * lda + kk;
    FragU u;
    u.q[0] = *(const uint4*)(p);
    u.q[1] = *(const uint4*)(p + 16);
    return u.v;
}

__device__ __forceinline__ v16bf load_b_frag(const bf16_t* Bm, int ldb,
                                             int n0, int k0, int lane) {
    // B is K x N fed from row-major W[N,K]: lane n (0-15) holds column n,
    // K = k0+[0..7] and k0+[16..23]; lanes 16-31 hold K = k0+[8..15] and k0+[24..31].
    int row = n0 + (lane & 15);
    int kk  = k0 + ((lane >> 4) << 3);
    const bf16_t* p = Bm + (size_t)row * ldb + kk;
    FragU u;
    u.q[0] = *(const uint4*)(p);
    u.q[1] = *(const uint4*)(p + 16);
    return u.v;
}

__device__ __forceinline__ void store_c_tile(float* C, int ldc,
                                             int m0, int n0, int lane, v8f acc) {
    // C 16x16 f32: VGPR r -> lanes 0-15: (M=r, N=lane); lanes 16-31: (M=r+8, N=lane-16)
    int col   = n0 + (lane & 15);
    int rbase = m0 + ((lane >> 4) << 3);
#pragma unroll
    for (int r = 0; r < 8; ++r)
        C[(size_t)(rbase + r) * ldc + col] = acc[r];
}

__global__ __launch_bounds__(256)
void wmma_gemm_bf16(const bf16_t* __restrict__ A,
                    const bf16_t* __restrict__ Bm,
                    float* __restrict__ C,
                    int M, int N, int K) {
    const int lane   = threadIdx.x & 31;
    const int wave   = threadIdx.x >> 5;          // 0..7
    const int wave_m = wave & 3;                  // 4 waves along M
    const int wave_n = wave >> 2;                 // 2 waves along N
    const int m0 = blockIdx.y * 128 + wave_m * 32;
    const int n0 = blockIdx.x * 128 + wave_n * 64;

    v8f acc00 = {}; v8f acc01 = {}; v8f acc02 = {}; v8f acc03 = {};
    v8f acc10 = {}; v8f acc11 = {}; v8f acc12 = {}; v8f acc13 = {};

    // stage 0: current fragments at k = 0
    v16bf a0 = load_a_frag(A, K, m0,      0, lane);
    v16bf a1 = load_a_frag(A, K, m0 + 16, 0, lane);
    v16bf b0 = load_b_frag(Bm, K, n0,      0, lane);
    v16bf b1 = load_b_frag(Bm, K, n0 + 16, 0, lane);
    v16bf b2 = load_b_frag(Bm, K, n0 + 32, 0, lane);
    v16bf b3 = load_b_frag(Bm, K, n0 + 48, 0, lane);

    for (int k = 0; k + 32 < K; k += 32) {
        const int kn = k + 32;
        // prefetch one stage beyond the register pipeline
        if (kn + 32 < K) {
            __builtin_prefetch(A + (size_t)(m0 + (lane & 15)) * K + kn + 32, 0, 3);
            __builtin_prefetch(Bm + (size_t)(n0 + (lane & 15)) * K + kn + 32, 0, 3);
        }
        // issue next-stage loads before consuming current fragments
        v16bf na0 = load_a_frag(A, K, m0,      kn, lane);
        v16bf na1 = load_a_frag(A, K, m0 + 16, kn, lane);
        v16bf nb0 = load_b_frag(Bm, K, n0,      kn, lane);
        v16bf nb1 = load_b_frag(Bm, K, n0 + 16, kn, lane);
        v16bf nb2 = load_b_frag(Bm, K, n0 + 32, kn, lane);
        v16bf nb3 = load_b_frag(Bm, K, n0 + 48, kn, lane);

        acc00 = __builtin_amdgcn_wmma_f32_16x16x32_bf16(false, a0, false, b0,
                                                        (short)0, acc00, false, false);
        acc10 = __builtin_amdgcn_wmma_f32_16x16x32_bf16(false, a1, false, b0,
                                                        (short)0, acc10, false, false);
        acc01 = __builtin_amdgcn_wmma_f32_16x16x32_bf16(false, a0, false, b1,
                                                        (short)0, acc01, false, false);
        acc11 = __builtin_amdgcn_wmma_f32_16x16x32_bf16(false, a1, false, b1,
                                                        (short)0, acc11, false, false);
        acc02 = __builtin_amdgcn_wmma_f32_16x16x32_bf16(false, a0, false, b2,
                                                        (short)0, acc02, false, false);
        acc12 = __builtin_amdgcn_wmma_f32_16x16x32_bf16(false, a1, false, b2,
                                                        (short)0, acc12, false, false);
        acc03 = __builtin_amdgcn_wmma_f32_16x16x32_bf16(false, a0, false, b3,
                                                        (short)0, acc03, false, false);
        acc13 = __builtin_amdgcn_wmma_f32_16x16x32_bf16(false, a1, false, b3,
                                                        (short)0, acc13, false, false);

        a0 = na0; a1 = na1; b0 = nb0; b1 = nb1; b2 = nb2; b3 = nb3;
    }

    // drain: last k-step
    acc00 = __builtin_amdgcn_wmma_f32_16x16x32_bf16(false, a0, false, b0,
                                                    (short)0, acc00, false, false);
    acc10 = __builtin_amdgcn_wmma_f32_16x16x32_bf16(false, a1, false, b0,
                                                    (short)0, acc10, false, false);
    acc01 = __builtin_amdgcn_wmma_f32_16x16x32_bf16(false, a0, false, b1,
                                                    (short)0, acc01, false, false);
    acc11 = __builtin_amdgcn_wmma_f32_16x16x32_bf16(false, a1, false, b1,
                                                    (short)0, acc11, false, false);
    acc02 = __builtin_amdgcn_wmma_f32_16x16x32_bf16(false, a0, false, b2,
                                                    (short)0, acc02, false, false);
    acc12 = __builtin_amdgcn_wmma_f32_16x16x32_bf16(false, a1, false, b2,
                                                    (short)0, acc12, false, false);
    acc03 = __builtin_amdgcn_wmma_f32_16x16x32_bf16(false, a0, false, b3,
                                                    (short)0, acc03, false, false);
    acc13 = __builtin_amdgcn_wmma_f32_16x16x32_bf16(false, a1, false, b3,
                                                    (short)0, acc13, false, false);

    store_c_tile(C, N, m0,      n0,      lane, acc00);
    store_c_tile(C, N, m0,      n0 + 16, lane, acc01);
    store_c_tile(C, N, m0,      n0 + 32, lane, acc02);
    store_c_tile(C, N, m0,      n0 + 48, lane, acc03);
    store_c_tile(C, N, m0 + 16, n0,      lane, acc10);
    store_c_tile(C, N, m0 + 16, n0 + 16, lane, acc11);
    store_c_tile(C, N, m0 + 16, n0 + 32, lane, acc12);
    store_c_tile(C, N, m0 + 16, n0 + 48, lane, acc13);
}

// ---------------------------------------------------------------------------
// Depthwise causal conv (width 4) + bias + SiLU on x_inner half of xz.
// xz: [NROWS, 4096] f32; out xc: [NROWS, 2048] f32
// ---------------------------------------------------------------------------
__device__ __forceinline__ float silu(float v) {
    return v / (1.0f + __expf(-v));
}

__global__ void conv_silu_kernel(const float* __restrict__ xz,
                                 const float* __restrict__ conv_w,
                                 const float* __restrict__ conv_b,
                                 float* __restrict__ xc) {
    int idx = blockIdx.x * blockDim.x + threadIdx.x;      // over BSZ*LSEQ*DINNER
    int d = idx & (DINNER - 1);
    int l = (idx >> 11) & (LSEQ - 1);
    int b = idx >> 23;
    float acc = conv_b[d];
#pragma unroll
    for (int j = 0; j < DCONV; ++j) {
        int lp = l - (DCONV - 1) + j;
        if (lp >= 0) {
            float xv = xz[((size_t)(b * LSEQ + lp)) * N_XZ + d];
            acc = fmaf(conv_w[d * DCONV + j], xv, acc);
        }
    }
    xc[(size_t)(b * LSEQ + l) * DINNER + d] = silu(acc);
}

// ---------------------------------------------------------------------------
// ssm = xc @ W_x^T  :  [NROWS, 33]  (tiny-N GEMM, W_x lives in L2: 264 KB)
// one block per row, threads 0..32 each compute one K=2048 dot product.
// ---------------------------------------------------------------------------
__global__ void ssm_proj_kernel(const float* __restrict__ xc,
                                const float* __restrict__ Wx,
                                float* __restrict__ ssm) {
    int row = blockIdx.x;
    int j   = threadIdx.x;
    if (j >= N_SSM) return;
    const float* xr = xc + (size_t)row * DINNER;
    const float* wr = Wx + (size_t)j * DINNER;
    float s = 0.f;
    for (int d = 0; d < DINNER; d += 4) {
        float4 xv = *(const float4*)(xr + d);
        float4 wv = *(const float4*)(wr + d);
        s = fmaf(xv.x, wv.x, s);
        s = fmaf(xv.y, wv.y, s);
        s = fmaf(xv.z, wv.z, s);
        s = fmaf(xv.w, wv.w, s);
    }
    ssm[(size_t)row * N_SSM + j] = s;
}

// ---------------------------------------------------------------------------
// Selective scan, wave32 layout: 16 lanes = 16 states of one (b,d) channel,
// 2 channels per wave. Butterfly shfl_xor(width 16) reduces y = sum_s h*C.
// Fuses softplus(dt), D*x skip, SiLU(z) gating; writes gated y as bf16.
// ---------------------------------------------------------------------------
__global__ __launch_bounds__(256)
void scan_kernel(const float* __restrict__ xc,
                 const float* __restrict__ ssm,
                 const float* __restrict__ xz,     // for z half
                 const float* __restrict__ A_log,
                 const float* __restrict__ Dv,
                 bf16_t* __restrict__ ybf) {
    int gtid = blockIdx.x * blockDim.x + threadIdx.x;
    int c = gtid >> 4;                 // channel index in [0, BSZ*DINNER)
    int s = gtid & 15;                 // state index
    int b = c >> 11;                   // c / DINNER
    int d = c & (DINNER - 1);

    float As = -__expf(A_log[d * DSTATE + s]);
    float Dd = Dv[d];
    float h  = 0.f;

    for (int t = 0; t < LSEQ; ++t) {
        size_t row = (size_t)(b * LSEQ + t);
        float xv  = xc[row * DINNER + d];
        const float* sr = ssm + row * N_SSM;
        float dtraw = sr[2 * DSTATE];
        // softplus
        float dt = (dtraw > 20.f) ? dtraw : log1pf(__expf(dtraw));
        float Bv = sr[s];
        float Cv = sr[DSTATE + s];

        h = __expf(dt * As) * h + dt * Bv * xv;
        float yp = h * Cv;
        // butterfly sum over the 16 states (stays within half-wave)
#pragma unroll
        for (int m = 8; m >= 1; m >>= 1)
            yp += __shfl_xor(yp, m, 16);

        if (s == 0) {
            float y  = yp + Dd * xv;
            float zv = xz[row * N_XZ + DINNER + d];
            ybf[row * DINNER + d] = (bf16_t)(y * silu(zv));
        }
    }
}

// ---------------------------------------------------------------------------
// Host-side orchestration
// ---------------------------------------------------------------------------
static inline size_t align256(size_t v) { return (v + 255) & ~(size_t)255; }

extern "C" void kernel_launch(void* const* d_in, const int* in_sizes, int n_in,
                              void* d_out, int out_size, void* d_ws, size_t ws_size,
                              hipStream_t stream) {
    const float* x      = (const float*)d_in[0];   // [4,4096,1024]
    const float* W_in   = (const float*)d_in[1];   // [4096,1024]
    const float* conv_w = (const float*)d_in[2];   // [2048,1,4]
    const float* conv_b = (const float*)d_in[3];   // [2048]
    const float* W_x    = (const float*)d_in[4];   // [33,2048]
    const float* A_log  = (const float*)d_in[5];   // [2048,16]
    const float* Dvec   = (const float*)d_in[6];   // [2048]
    const float* W_out  = (const float*)d_in[7];   // [1024,2048]
    float* out = (float*)d_out;                    // [4,4096,1024]

    // workspace carve-up
    char* ws = (char*)d_ws;
    size_t off = 0;
    bf16_t* x_bf    = (bf16_t*)(ws + off); off += align256((size_t)NROWS * DMODEL * 2);
    bf16_t* Win_bf  = (bf16_t*)(ws + off); off += align256((size_t)N_XZ  * DMODEL * 2);
    bf16_t* Wout_bf = (bf16_t*)(ws + off); off += align256((size_t)DMODEL * DINNER * 2);
    bf16_t* y_bf    = (bf16_t*)(ws + off); off += align256((size_t)NROWS * DINNER * 2);
    float*  xz      = (float*)(ws + off);  off += align256((size_t)NROWS * N_XZ * 4);
    float*  xc      = (float*)(ws + off);  off += align256((size_t)NROWS * DINNER * 4);
    float*  ssm     = (float*)(ws + off);  off += align256((size_t)NROWS * N_SSM * 4);
    (void)ws_size;

    // 1) fp32 -> bf16 conversions for GEMM operands
    {
        int n = NROWS * DMODEL;
        cvt_f32_bf16<<<(n + 255) / 256, 256, 0, stream>>>(x, x_bf, n);
        n = N_XZ * DMODEL;
        cvt_f32_bf16<<<(n + 255) / 256, 256, 0, stream>>>(W_in, Win_bf, n);
        n = DMODEL * DINNER;
        cvt_f32_bf16<<<(n + 255) / 256, 256, 0, stream>>>(W_out, Wout_bf, n);
    }

    // 2) xz = x @ W_in^T   (M=16384, N=4096, K=1024)
    wmma_gemm_bf16<<<dim3(N_XZ / 128, NROWS / 128), 256, 0, stream>>>(
        x_bf, Win_bf, xz, NROWS, N_XZ, DMODEL);

    // 3) depthwise conv + SiLU  -> xc
    {
        long long n = (long long)NROWS * DINNER;
        conv_silu_kernel<<<(int)(n / 256), 256, 0, stream>>>(xz, conv_w, conv_b, xc);
    }

    // 4) ssm = xc @ W_x^T  (N=33)
    ssm_proj_kernel<<<NROWS, 64, 0, stream>>>(xc, W_x, ssm);

    // 5) selective scan + gating -> y_bf
    {
        int total = BSZ * DINNER * DSTATE;   // 131072 threads
        scan_kernel<<<total / 256, 256, 0, stream>>>(xc, ssm, xz, A_log, Dvec, y_bf);
    }

    // 6) out = y_gated @ W_out^T   (M=16384, N=1024, K=2048)
    wmma_gemm_bf16<<<dim3(DMODEL / 128, NROWS / 128), 256, 0, stream>>>(
        y_bf, Wout_bf, out, NROWS, DMODEL, DINNER);
}
